// GNN_15281493639468
// MI455X (gfx1250) — matrix operand order repaired and checked by reading, not compiled, not run on hardware
//
#include <hip/hip_runtime.h>

typedef __attribute__((ext_vector_type(16))) _Float16 v16h;
typedef __attribute__((ext_vector_type(8)))  _Float16 v8h;
typedef __attribute__((ext_vector_type(8)))  float    v8f;

#define GCN_L   5
#define GCN_EMB 256
#define GCN_HID 32
#define GCN_OUT 32

// ---------------------------------------------------------------- utilities

__global__ __launch_bounds__(256) void gcn_f32_to_f16(const float* __restrict__ in,
                                                      _Float16* __restrict__ out, int n) {
    for (int i = blockIdx.x * blockDim.x + threadIdx.x; i < n; i += gridDim.x * blockDim.x)
        out[i] = (_Float16)in[i];
}

// Transpose+convert weights: W1t0 [32][256], W1tR 4x[32][32], W2t 5x[32][32]
__global__ __launch_bounds__(256) void gcn_prep_weights(const float* __restrict__ W1_0,
                                                        const float* __restrict__ W1_rest,
                                                        const float* __restrict__ W2,
                                                        _Float16* __restrict__ W1t0,
                                                        _Float16* __restrict__ W1tR,
                                                        _Float16* __restrict__ W2t) {
    int i = blockIdx.x * blockDim.x + threadIdx.x;
    if (i < 32 * 256) {                     // [c][k] <- W1_0[k][c]
        int c = i >> 8, k = i & 255;
        W1t0[i] = (_Float16)W1_0[k * 32 + c];
    }
    if (i < 4 * 32 * 32) {                  // per layer [c][k] <- W1_rest[l][k][c]
        int l = i >> 10, r = i & 1023, c = r >> 5, k = r & 31;
        W1tR[i] = (_Float16)W1_rest[l * 1024 + k * 32 + c];
    }
    if (i < 5 * 32 * 32) {                  // per layer [c][k] <- W2[l][k][c]
        int l = i >> 10, r = i & 1023, c = r >> 5, k = r & 31;
        W2t[i] = (_Float16)W2[l * 1024 + k * 32 + c];
    }
}

__global__ void gcn_zero_stats(float* stats) {
    if (threadIdx.x < 64) stats[threadIdx.x] = 0.0f;
}

// ------------------------------------------------------- fused MLP (WMMA)
// z = relu(h@W1 + b1) @ W2 + b2 ; agg = z + bias  (self-loop + bias fused)
// 8 waves/block, 16 rows/wave, WMMA 16x16x32 f16->f32.
__global__ __launch_bounds__(256)
void gcn_mlp_wmma(const _Float16* __restrict__ hf16, int Din,
                  const _Float16* __restrict__ W1t,   // [32][Din]
                  const float*    __restrict__ b1l,   // [32]
                  const _Float16* __restrict__ W2t,   // [32][32]
                  const float*    __restrict__ b2l,   // [32]
                  const float*    __restrict__ biasl, // [32]
                  float* __restrict__ z, float* __restrict__ agg, int N) {
    __shared__ _Float16 lsh[8 * 16 * 32];   // per-wave 16x32 H1 staging

    const int tid  = threadIdx.x;
    const int wid  = tid >> 5;
    const int lane = tid & 31;
    const int l15  = lane & 15;
    const int hi   = lane >> 4;             // 0: K-low half, 1: K-high half
    const int rowBase = blockIdx.x * 128 + wid * 16;

    int rowA  = rowBase + l15;
    int rowAc = rowA < N ? rowA : (N - 1);  // clamp: keep EXEC all-ones for WMMA

    // ---- GEMM1: acc[n] = h(16xDin) @ W1(Dinx16) for n-th 16-col tile
    const _Float16* arow = hf16 + (size_t)rowAc * Din + hi * 16;
    const _Float16* w1c0 = W1t + (size_t)(l15)      * Din + hi * 16;
    const _Float16* w1c1 = W1t + (size_t)(16 + l15) * Din + hi * 16;

    v8f acc0 = {}; v8f acc1 = {};
    for (int k0 = 0; k0 < Din; k0 += 32) {
        // A layout: lane<16 K=0..15, lane>=16 K=16..31 (matches B)
        v8h alo = *(const v8h*)(arow + k0);
        v8h ahi = *(const v8h*)(arow + k0 + 8);
        v16h a = __builtin_shufflevector(alo, ahi, 0,1,2,3,4,5,6,7,8,9,10,11,12,13,14,15);
        // B layout: lane<16 col=lane K=0..15 ; lane>=16 col=lane-16 K=16..31
        v16h bb0 = *(const v16h*)(w1c0 + k0);
        v16h bb1 = *(const v16h*)(w1c1 + k0);
        acc0 = __builtin_amdgcn_wmma_f32_16x16x32_f16(false, a, false, bb0, (short)0, acc0, false, false);
        acc1 = __builtin_amdgcn_wmma_f32_16x16x32_f16(false, a, false, bb1, (short)0, acc1, false, false);
    }

    // ---- relu + b1, D-layout -> LDS (row-major 16x32 f16 per wave)
    const float bc0 = b1l[l15], bc1 = b1l[16 + l15];
    _Float16* ws = lsh + wid * 512;
#pragma unroll
    for (int r = 0; r < 8; ++r) {
        int rrow = r + hi * 8;              // D layout: lanes>=16 hold M=r+8
        ws[rrow * 32 + l15]      = (_Float16)fmaxf(acc0[r] + bc0, 0.0f);
        ws[rrow * 32 + 16 + l15] = (_Float16)fmaxf(acc1[r] + bc1, 0.0f);
    }
    __syncthreads();

    // ---- GEMM2: reload H1 in A layout, K=32
    const _Float16* a2p = ws + l15 * 32 + hi * 16;
    v8h a2lo = *(const v8h*)(a2p);
    v8h a2hi = *(const v8h*)(a2p + 8);
    v16h a2 = __builtin_shufflevector(a2lo, a2hi, 0,1,2,3,4,5,6,7,8,9,10,11,12,13,14,15);
    v16h w2b0 = *(const v16h*)(W2t + (size_t)(l15)      * 32 + hi * 16);
    v16h w2b1 = *(const v16h*)(W2t + (size_t)(16 + l15) * 32 + hi * 16);
    v8f z0 = {}; v8f z1 = {};
    z0 = __builtin_amdgcn_wmma_f32_16x16x32_f16(false, a2, false, w2b0, (short)0, z0, false, false);
    z1 = __builtin_amdgcn_wmma_f32_16x16x32_f16(false, a2, false, w2b1, (short)0, z1, false, false);

    // ---- epilogue: z = .. + b2 ; agg = z + bias (self loop init)
    const float b20 = b2l[l15], b21 = b2l[16 + l15];
    const float bi0 = biasl[l15], bi1 = biasl[16 + l15];
#pragma unroll
    for (int r = 0; r < 8; ++r) {
        int grow = rowBase + r + hi * 8;
        if (grow < N) {
            float zv0 = z0[r] + b20, zv1 = z1[r] + b21;
            size_t o = (size_t)grow * 32;
            z[o + l15]        = zv0;
            z[o + 16 + l15]   = zv1;
            agg[o + l15]      = zv0 + bi0;
            agg[o + 16 + l15] = zv1 + bi1;
        }
    }
}

// ---------------------------------------------------------- edge scatter
// one wave per edge, lane = channel; edge weights are all 1.0 (deg==1)
__global__ __launch_bounds__(256)
void gcn_scatter(const int* __restrict__ src, const int* __restrict__ dst,
                 const float* __restrict__ z, float* __restrict__ agg, int E) {
    int e = blockIdx.x * 8 + (threadIdx.x >> 5);
    if (e >= E) return;
    int lane = threadIdx.x & 31;
    int s = src[e], d = dst[e];
    float v = z[(size_t)s * 32 + lane];
    __hip_atomic_fetch_add(&agg[(size_t)d * 32 + lane], v,
                           __ATOMIC_RELAXED, __HIP_MEMORY_SCOPE_AGENT);
}

// ------------------------------------------------------------- batchnorm
__global__ __launch_bounds__(256)
void gcn_bn_stats(const float* __restrict__ agg, float* __restrict__ stats, int N) {
    __shared__ float ss[256], sq[256];
    int tid = threadIdx.x;
    float s = 0.0f, q = 0.0f;
    int total = N * 32, stride = gridDim.x * blockDim.x;  // stride % 32 == 0
    for (int i = blockIdx.x * blockDim.x + tid; i < total; i += stride) {
        float v = agg[i];
        s += v; q += v * v;
    }
    ss[tid] = s; sq[tid] = q;
    __syncthreads();
    for (int off = 128; off >= 32; off >>= 1) {
        if (tid < off) { ss[tid] += ss[tid + off]; sq[tid] += sq[tid + off]; }
        __syncthreads();
    }
    if (tid < 32) {
        __hip_atomic_fetch_add(&stats[tid],      ss[tid], __ATOMIC_RELAXED, __HIP_MEMORY_SCOPE_AGENT);
        __hip_atomic_fetch_add(&stats[32 + tid], sq[tid], __ATOMIC_RELAXED, __HIP_MEMORY_SCOPE_AGENT);
    }
}

__global__ __launch_bounds__(256)
void gcn_bn_apply(const float* __restrict__ agg, const float* __restrict__ stats,
                  const float* __restrict__ gammal, const float* __restrict__ betal,
                  _Float16* __restrict__ hf16out, float* __restrict__ fout,
                  int N, int doRelu, int writeF32) {
    int total = N * 32, stride = gridDim.x * blockDim.x;
    float invN = 1.0f / (float)N;
    for (int i = blockIdx.x * blockDim.x + threadIdx.x; i < total; i += stride) {
        int c = i & 31;
        float mu  = stats[c] * invN;
        float var = stats[32 + c] * invN - mu * mu;
        float inv = rsqrtf(var + 1e-5f);
        float v = (agg[i] - mu) * inv * gammal[c] + betal[c];
        if (doRelu) v = fmaxf(v, 0.0f);
        if (writeF32) fout[i] = v;
        else          hf16out[i] = (_Float16)v;
    }
}

// ---------------------------------------------------------------- driver

extern "C" void kernel_launch(void* const* d_in, const int* in_sizes, int n_in,
                              void* d_out, int out_size, void* d_ws, size_t ws_size,
                              hipStream_t stream) {
    const float* x       = (const float*)d_in[0];
    const int*   ei      = (const int*)  d_in[1];
    const float* W1_0    = (const float*)d_in[2];
    const float* W1_rest = (const float*)d_in[3];
    const float* b1      = (const float*)d_in[4];
    const float* W2      = (const float*)d_in[5];
    const float* b2      = (const float*)d_in[6];
    const float* bias    = (const float*)d_in[7];
    const float* gamma   = (const float*)d_in[8];
    const float* beta    = (const float*)d_in[9];

    const int N = in_sizes[0] / GCN_EMB;
    const int E = in_sizes[1] / 2;
    const int* srcp = ei;
    const int* dstp = ei + E;

    // workspace layout (256B aligned)
    char* ws = (char*)d_ws;
    size_t off = 0;
    auto carve = [&](size_t bytes) -> char* {
        char* p = ws + off;
        off = (off + bytes + 255) & ~(size_t)255;
        return p;
    };
    _Float16* hf16  = (_Float16*)carve((size_t)N * GCN_EMB * sizeof(_Float16));
    float*    zbuf  = (float*)   carve((size_t)N * 32 * sizeof(float));
    float*    agg   = (float*)   carve((size_t)N * 32 * sizeof(float));
    float*    stats = (float*)   carve(64 * sizeof(float));
    _Float16* W1t0  = (_Float16*)carve(32 * 256 * sizeof(_Float16));
    _Float16* W1tR  = (_Float16*)carve(4 * 32 * 32 * sizeof(_Float16));
    _Float16* W2t   = (_Float16*)carve(5 * 32 * 32 * sizeof(_Float16));
    (void)ws_size; (void)n_in;

    // one-time prep
    {
        int n = N * GCN_EMB;
        gcn_f32_to_f16<<<(n + 255) / 256, 256, 0, stream>>>(x, hf16, n);
        gcn_prep_weights<<<32, 256, 0, stream>>>(W1_0, W1_rest, W2, W1t0, W1tR, W2t);
    }

    const int mlpBlocks  = (N + 127) / 128;
    const int scatBlocks = (E + 7) / 8;
    const int bnBlocks   = 512;
    const int elemBlocks = (N * 32 + 255) / 256;

    for (int l = 0; l < GCN_L; ++l) {
        const int Din = (l == 0) ? GCN_EMB : GCN_OUT;
        const _Float16* W1tl = (l == 0) ? W1t0 : (W1tR + (size_t)(l - 1) * 1024);
        gcn_mlp_wmma<<<mlpBlocks, 256, 0, stream>>>(
            hf16, Din, W1tl, b1 + l * 32, W2t + (size_t)l * 1024,
            b2 + l * 32, bias + l * 32, zbuf, agg, N);
        gcn_scatter<<<scatBlocks, 256, 0, stream>>>(srcp, dstp, zbuf, agg, E);
        gcn_zero_stats<<<1, 64, 0, stream>>>(stats);
        gcn_bn_stats<<<bnBlocks, 256, 0, stream>>>(agg, stats, N);
        const int last = (l == GCN_L - 1);
        gcn_bn_apply<<<elemBlocks, 256, 0, stream>>>(
            agg, stats, gamma + l * 32, beta + l * 32,
            hf16, (float*)d_out, N, /*doRelu=*/!last, /*writeF32=*/last);
    }
    (void)out_size;
}
